// Mamba2BonsaiMixer_22101901705779
// MI455X (gfx1250) — matrix operand order repaired and checked by reading, not compiled, not run on hardware
//
#include <hip/hip_runtime.h>
#include <hip/hip_bf16.h>
#include <math.h>

// ---------------------------------------------------------------------------
// Mamba2 mixer forward for MI455X (gfx1250), wave32, bf16 WMMA everywhere.
// GEMM operands are pre-packed once to bf16 in WMMA-fragment order in global
// memory; GEMM staging is then a pure contiguous copy (b128 load -> b128 LDS
// store), and each lane reads its fragment as one aligned v16bf.
// ---------------------------------------------------------------------------

typedef __attribute__((ext_vector_type(16))) __bf16 v16bf;
typedef __attribute__((ext_vector_type(4)))  __bf16 v4bf;
typedef __attribute__((ext_vector_type(8)))  float  v8f;

#define WMMA_BF16(A, B, C) \
  __builtin_amdgcn_wmma_f32_16x16x32_bf16(false, (A), false, (B), (short)0, (C), false, false)

// Model constants
#define N_HIDDEN  2048
#define N_HEADS   64
#define D_STATE   128
#define D_CONV    4
#define CHUNK     128
#define INNER     4096
#define HEAD_DIM  64
#define CONV_DIM  4352          // INNER + 2*D_STATE
#define D_IN_PROJ 8512          // 2*INNER + 2*D_STATE + N_HEADS
#define WIN_NPAD  8576          // D_IN_PROJ rounded up to 128
#define BATCH     2
#define SEQ       2048
#define NTOK      (BATCH * SEQ) // 4096
#define NCHUNK    (SEQ / CHUNK) // 16
#define DT_MIN    0.001f
#define DT_MAX    0.1f

// ---------------------------------------------------------------------------
// Fragment-order helpers (CDNA5 WMMA 16-bit layouts).
// A-frag (16x32): lane = (m&15) + 16*((kk>>3)&1), elem = (kk&7) + 8*(kk>>4)
// B-frag (32x16): lane = (n&15) + 16*((kk>>3)&1), elem = (kk&7) + 8*(kk>>4)
// Buffers indexed [tile_block][k_slab][lane] of v16bf.
// ---------------------------------------------------------------------------
static __device__ __forceinline__ void stage_a(__bf16* buf, int m, int k,
                                               float v, int kslabs) {
  int kk = k & 31, ks = k >> 5;
  int lane = (m & 15) + 16 * ((kk >> 3) & 1);
  int e = (kk & 7) + 8 * (kk >> 4);
  buf[((((m >> 4) * kslabs) + ks) * 32 + lane) * 16 + e] = (__bf16)v;
}
static __device__ __forceinline__ void stage_b(__bf16* buf, int k, int n,
                                               float v, int kslabs) {
  int kk = k & 31, ks = k >> 5;
  int lane = (n & 15) + 16 * ((kk >> 3) & 1);
  int e = (kk & 7) + 8 * (kk >> 4);
  buf[((((n >> 4) * kslabs) + ks) * 32 + lane) * 16 + e] = (__bf16)v;
}
static __device__ __forceinline__ void stage_a4(__bf16* buf, int m, int k4,
                                                float4 fv, int kslabs) {
  int kk = k4 & 31, ks = k4 >> 5;
  int lane = (m & 15) + 16 * ((kk >> 3) & 1);
  int e = (kk & 7) + 8 * (kk >> 4);
  v4bf v;
  v[0] = (__bf16)fv.x; v[1] = (__bf16)fv.y; v[2] = (__bf16)fv.z; v[3] = (__bf16)fv.w;
  *(v4bf*)(buf + ((((m >> 4) * kslabs) + ks) * 32 + lane) * 16 + e) = v;
}
static __device__ __forceinline__ void stage_b4(__bf16* buf, int k4, int n,
                                                float4 fv, int kslabs) {
  int kk = k4 & 31, ks = k4 >> 5;
  int lane = (n & 15) + 16 * ((kk >> 3) & 1);
  int e = (kk & 7) + 8 * (kk >> 4);
  v4bf v;
  v[0] = (__bf16)fv.x; v[1] = (__bf16)fv.y; v[2] = (__bf16)fv.z; v[3] = (__bf16)fv.w;
  *(v4bf*)(buf + ((((n >> 4) * kslabs) + ks) * 32 + lane) * 16 + e) = v;
}

static __device__ __forceinline__ float silu_f(float x) {
  return x / (1.f + expf(-x));
}

// ---------------------------------------------------------------------------
// Pack fp32 row-major A[R][K] -> bf16 A-fragment order [R/16][K/32][32][16].
// grid = (K/32, R/128), block = 256.
// ---------------------------------------------------------------------------
template <int R, int K>
__global__ __launch_bounds__(256) void pack_a_kernel(
    const float* __restrict__ A, __bf16* __restrict__ out)
{
  __shared__ v16bf sT[8][32];
  const int tid = threadIdx.x;
  const int ks = blockIdx.x;
  const int r0 = blockIdx.y * 128;
  __bf16* pT = (__bf16*)sT;
  const int m = tid >> 3, k4 = (tid & 7) << 2;
#pragma unroll
  for (int u = 0; u < 4; ++u) {
    float4 fv = *(const float4*)(A + (size_t)(r0 + m + 32 * u) * K + ks * 32 + k4);
    stage_a4(pT, m + 32 * u, k4, fv, 1);
  }
  __syncthreads();
  const int blk = tid >> 5, lane = tid & 31;
  size_t o = ((size_t)((r0 >> 4) + blk) * (K / 32) + ks) * 512 + lane * 16;
  *(v16bf*)(out + o) = sT[blk][lane];
}

// ---------------------------------------------------------------------------
// Pack fp32 row-major B[K][N] -> bf16 B-fragment order [NPAD/16][K/32][32][16]
// (zero-padded to NPAD columns).  grid = (NPAD/128, K/32), block = 256.
// ---------------------------------------------------------------------------
template <int K, int N, int NPAD>
__global__ __launch_bounds__(256) void pack_b_kernel(
    const float* __restrict__ B, __bf16* __restrict__ out)
{
  __shared__ v16bf sT[8][32];
  const int tid = threadIdx.x;
  const int n0 = blockIdx.x * 128;
  const int ks = blockIdx.y;
  __bf16* pT = (__bf16*)sT;
  const int k = tid >> 5, n4 = (tid & 31) << 2;
  const int gcol = n0 + n4;
  const bool ok = (N % 128 == 0) || (gcol < N);
  const int gc = ok ? gcol : (N - 4);
#pragma unroll
  for (int u = 0; u < 4; ++u) {
    float4 fv = *(const float4*)(B + (size_t)(ks * 32 + k + 8 * u) * N + gc);
    if (!ok) fv = make_float4(0.f, 0.f, 0.f, 0.f);
    float vb[4] = {fv.x, fv.y, fv.z, fv.w};
#pragma unroll
    for (int w = 0; w < 4; ++w) stage_b(pT, k + 8 * u, n4 + w, vb[w], 1);
  }
  __syncthreads();
  const int blk = tid >> 5, lane = tid & 31;
  size_t o = ((size_t)((n0 >> 4) + blk) * (K / 32) + ks) * 512 + lane * 16;
  *(v16bf*)(out + o) = sT[blk][lane];
}

// ---------------------------------------------------------------------------
// Packed-operand GEMM: C[M,N] = A*B + bias, bf16 WMMA, double-buffered LDS.
// Staging = contiguous copy: per thread per buffer one 32B read + 32B write.
// Workgroup tile 128x128, 8 waves (4x2), wave tile 32x64 (2x4 WMMA tiles).
// ---------------------------------------------------------------------------
template <int M, int N, int NPAD, int K>
__global__ __launch_bounds__(256) void gemm_pk(
    const __bf16* __restrict__ Ap, const __bf16* __restrict__ Bp,
    const float* __restrict__ bias, float* __restrict__ C)
{
  constexpr int KS = K / 32;
  __shared__ v16bf sAf[2][8][32];   // [buf][row-tile][lane]
  __shared__ v16bf sBf[2][8][32];   // [buf][col-tile][lane]

  const int tid  = threadIdx.x;
  const int wid  = tid >> 5, lane = tid & 31;
  const int g    = lane >> 4, r   = lane & 15;
  const int wm   = wid & 3,  wn   = wid >> 2;
  const int bM   = blockIdx.y * 128, bN = blockIdx.x * 128;

  // Contiguous staging sources: block `wid` of this tile, 32B per lane.
  const __bf16* aSrc = Ap + ((size_t)((bM >> 4) + wid) * KS) * 512 + lane * 16;
  const __bf16* bSrc = Bp + ((size_t)((bN >> 4) + wid) * KS) * 512 + lane * 16;

  v8f acc[2][4] = {};

  auto stage = [&](int buf, int ks) {
    sAf[buf][wid][lane] = *(const v16bf*)(aSrc + (size_t)ks * 512);
    sBf[buf][wid][lane] = *(const v16bf*)(bSrc + (size_t)ks * 512);
  };

  stage(0, 0);
  __syncthreads();
  int cur = 0;
  for (int ks = 0; ks < KS; ++ks) {
    if (ks + 1 < KS) stage(cur ^ 1, ks + 1);   // overlap next-tile staging
    v16bf af[2], bfr[4];
#pragma unroll
    for (int i = 0; i < 2; ++i) af[i] = sAf[cur][wm * 2 + i][lane];
#pragma unroll
    for (int j = 0; j < 4; ++j) bfr[j] = sBf[cur][wn * 4 + j][lane];
#pragma unroll
    for (int i = 0; i < 2; ++i)
#pragma unroll
      for (int j = 0; j < 4; ++j)
        acc[i][j] = WMMA_BF16(af[i], bfr[j], acc[i][j]);
    __syncthreads();
    cur ^= 1;
  }

#pragma unroll
  for (int i = 0; i < 2; ++i)
#pragma unroll
    for (int j = 0; j < 4; ++j)
#pragma unroll
      for (int q = 0; q < 8; ++q) {
        int row = bM + wm * 32 + i * 16 + q + 8 * g;
        int col = bN + wn * 64 + j * 16 + r;
        if ((N % 128 == 0) || col < N)
          C[(size_t)row * N + col] = acc[i][j][q] + bias[col];
      }
}

// ---------------------------------------------------------------------------
// dt = clip(softplus(dt_raw + dt_bias), DT_MIN, DT_MAX);  a_disc = -exp(A_log)*dt
// ---------------------------------------------------------------------------
__global__ __launch_bounds__(256) void dt_prep(
    const float* __restrict__ zxbcdt, const float* __restrict__ A_log,
    const float* __restrict__ dt_bias, float* __restrict__ dtw,
    float* __restrict__ adisc)
{
  int idx = blockIdx.x * 256 + threadIdx.x;   // over NTOK * N_HEADS
  int h = idx & (N_HEADS - 1);
  int token = idx >> 6;
  float raw = zxbcdt[(size_t)token * D_IN_PROJ + (D_IN_PROJ - N_HEADS) + h] + dt_bias[h];
  float sp = (raw > 20.f) ? raw : log1pf(expf(raw));
  float dtv = fminf(fmaxf(sp, DT_MIN), DT_MAX);
  dtw[idx] = dtv;
  adisc[idx] = -expf(A_log[h]) * dtv;
}

// ---------------------------------------------------------------------------
// Depthwise causal conv (4 taps) + bias + SiLU.  grid = (CONV_DIM/256, NTOK).
// ---------------------------------------------------------------------------
__global__ __launch_bounds__(256) void conv_silu(
    const float* __restrict__ zxbcdt, const float* __restrict__ conv_w,
    const float* __restrict__ conv_b, float* __restrict__ xbc)
{
  int c = blockIdx.x * 256 + threadIdx.x;
  int token = blockIdx.y;
  int l = token & (SEQ - 1);
  int b = token >> 11;               // SEQ = 2048
  float acc = conv_b[c];
#pragma unroll
  for (int k = 0; k < D_CONV; ++k) {
    int lp = l + k - (D_CONV - 1);
    if (lp >= 0)
      acc += conv_w[k * CONV_DIM + c] *
             zxbcdt[(size_t)(b * SEQ + lp) * D_IN_PROJ + INNER + c];
  }
  xbc[(size_t)token * CONV_DIM + c] = silu_f(acc);
}

// ---------------------------------------------------------------------------
// SSD per (b, chunk, head): y_diag = (C B^T ∘ L) Xdt,  local_states = Xdec^T B
// ---------------------------------------------------------------------------
__global__ __launch_bounds__(256) void ssd_chunk1(
    const float* __restrict__ xbc, const float* __restrict__ dtw,
    const float* __restrict__ adisc, float* __restrict__ y,
    float* __restrict__ states, float* __restrict__ acum_ws,
    float* __restrict__ asum_ws)
{
  __shared__ v16bf sCa[8][4][32];    // C as A-frags (phase G)
  __shared__ v16bf sBg[8][4][32];    // B^T as B-frags (phase G); reused as M A-frags
  __shared__ v16bf sBs[8][4][32];    // B as B-frags (phase S)
  __shared__ v16bf sXd[4][4][32];    // x*dt as B-frags (phase Y)
  __shared__ v16bf sXdec[4][4][32];  // (x*dt*decay)^T as A-frags (phase S)
  __shared__ float sAcum[128], sDt[128], sAd[128];

  const int tid = threadIdx.x;
  const int bx  = blockIdx.x;
  const int h   = bx & 63;
  const int c   = (bx >> 6) % NCHUNK;
  const int b   = bx / (64 * NCHUNK);
  const int l0  = c * CHUNK;
  const float* xc = xbc + (size_t)(b * SEQ + l0) * CONV_DIM;
  __bf16* pCa = (__bf16*)sCa;
  __bf16* pBg = (__bf16*)sBg;
  __bf16* pBs = (__bf16*)sBs;
  __bf16* pXd = (__bf16*)sXd;
  __bf16* pXe = (__bf16*)sXdec;

  if (tid < 128) {
    size_t ti = (size_t)(b * SEQ + l0 + tid) * N_HEADS + h;
    sDt[tid] = dtw[ti];
    sAd[tid] = adisc[ti];
  }
  // stage C and B (128x128 each, float4 reads; l = (tid>>5) + 8u)
  {
    const int lB = tid >> 5, n4 = (tid & 31) << 2;
#pragma unroll
    for (int u = 0; u < 16; ++u) {
      int l = lB + 8 * u;
      float4 cv = *(const float4*)(xc + (size_t)l * CONV_DIM + (INNER + D_STATE) + n4);
      float4 bv = *(const float4*)(xc + (size_t)l * CONV_DIM + INNER + n4);
      stage_a4(pCa, l, n4, cv, 4);               // A rows=l, k=n (packed)
      stage_b4(pBg, n4, l, bv, 4);               // B-frag k=n, col=l (packed)
      float ba[4] = {bv.x, bv.y, bv.z, bv.w};
#pragma unroll
      for (int w = 0; w < 4; ++w)
        stage_b(pBs, l, n4 + w, ba[w], 4);       // B-frag k=l, col=n (scattered)
    }
  }
  __syncthreads();
  if (tid == 0) {
    float s = 0.f;
    for (int l = 0; l < 128; ++l) { s += sAd[l]; sAcum[l] = s; }
  }
  __syncthreads();
  float alast = sAcum[127];
  // stage Xd / Xdec (128 x 64, float4 reads; s = (tid>>4) + 16u)
  {
    const int sB = tid >> 4, p4 = (tid & 15) << 2;
#pragma unroll
    for (int u = 0; u < 8; ++u) {
      int s_ = sB + 16 * u;
      float4 xv = *(const float4*)(xc + (size_t)s_ * CONV_DIM + h * HEAD_DIM + p4);
      float dtv = sDt[s_];
      float dec = expf(alast - sAcum[s_]);
      float xa[4] = {xv.x, xv.y, xv.z, xv.w};
#pragma unroll
      for (int w = 0; w < 4; ++w) {
        float xd = xa[w] * dtv;
        stage_b(pXd, s_, p4 + w, xd, 4);          // B-frag k=s, col=p
        stage_a(pXe, p4 + w, s_, xd * dec, 4);    // A rows=p, k=l
      }
    }
  }
  if (tid < 128)
    acum_ws[(((size_t)(b * NCHUNK + c)) * N_HEADS + h) * 128 + tid] = sAcum[tid];
  if (tid == 0)
    asum_ws[(b * NCHUNK + c) * N_HEADS + h] = alast;
  __syncthreads();

  const int wid = tid >> 5, lane = tid & 31;
  const int g = lane >> 4, r = lane & 15;

  // Phase G: G[l,s] = sum_n C[l,n] * B[s,n]  (wave wid owns tile row wid)
  v8f accG[8] = {};
#pragma unroll
  for (int ko = 0; ko < 4; ++ko) {
    v16bf af = sCa[wid][ko][lane];
#pragma unroll
    for (int j = 0; j < 8; ++j)
      accG[j] = WMMA_BF16(af, sBg[j][ko][lane], accG[j]);
  }
  __syncthreads();

  // Apply causal decay mask L and restage as M A-frags into sBg.
#pragma unroll
  for (int j = 0; j < 8; ++j)
#pragma unroll
    for (int q = 0; q < 8; ++q) {
      int l = wid * 16 + q + 8 * g;
      int s = j * 16 + r;
      float v = (l >= s) ? accG[j][q] * expf(sAcum[l] - sAcum[s]) : 0.f;
      stage_a(pBg, l, s, v, 4);
    }
  __syncthreads();

  // Phase Y: y_diag = M @ Xd   (128x64, contract s=128)
  {
    v8f accY[4] = {};
#pragma unroll
    for (int ko = 0; ko < 4; ++ko) {
      v16bf af = sBg[wid][ko][lane];
#pragma unroll
      for (int j = 0; j < 4; ++j)
        accY[j] = WMMA_BF16(af, sXd[j][ko][lane], accY[j]);
    }
#pragma unroll
    for (int j = 0; j < 4; ++j)
#pragma unroll
      for (int q = 0; q < 8; ++q) {
        int l = wid * 16 + q + 8 * g;
        int p = j * 16 + r;
        y[(size_t)(b * SEQ + l0 + l) * INNER + h * HEAD_DIM + p] = accY[j][q];
      }
  }

  // Phase S: states[p,n] = sum_l Xdec[l,p] * B[l,n]  (64x128, contract l=128)
  {
    const int pr = wid & 3, cg = wid >> 2;
    v8f accS[4] = {};
#pragma unroll
    for (int ko = 0; ko < 4; ++ko) {
      v16bf af = sXdec[pr][ko][lane];
#pragma unroll
      for (int j = 0; j < 4; ++j)
        accS[j] = WMMA_BF16(af, sBs[cg * 4 + j][ko][lane], accS[j]);
    }
    size_t sbase = (((size_t)(b * NCHUNK + c)) * N_HEADS + h) * (HEAD_DIM * D_STATE);
#pragma unroll
    for (int j = 0; j < 4; ++j)
#pragma unroll
      for (int q = 0; q < 8; ++q) {
        int p = pr * 16 + q + 8 * g;
        int n = cg * 64 + j * 16 + r;
        states[sbase + (size_t)p * D_STATE + n] = accS[j][q];
      }
  }
}

// ---------------------------------------------------------------------------
// Inter-chunk state scan (in-place): buf[c] <- S_c ; S <- exp(asum_c)*S + local_c
// ---------------------------------------------------------------------------
__global__ __launch_bounds__(256) void scan_states(
    float* __restrict__ states, const float* __restrict__ asum)
{
  const int tid = threadIdx.x;
  const int bh = blockIdx.x;
  const int b = bh >> 6, h = bh & 63;
  float S[32];
#pragma unroll
  for (int j = 0; j < 32; ++j) S[j] = 0.f;
  for (int c = 0; c < NCHUNK; ++c) {
    size_t base = (((size_t)(b * NCHUNK + c)) * N_HEADS + h) * (HEAD_DIM * D_STATE);
    float dec = expf(asum[(b * NCHUNK + c) * N_HEADS + h]);
#pragma unroll
    for (int j = 0; j < 32; ++j) {
      size_t idx = base + tid + 256 * j;
      float loc = states[idx];
      states[idx] = S[j];
      S[j] = dec * S[j] + loc;
    }
  }
}

// ---------------------------------------------------------------------------
// y += (exp(a_cum) ∘ C) @ S^T + D*x     (128x64 per (b,chunk,head), WMMA)
// ---------------------------------------------------------------------------
__global__ __launch_bounds__(256) void ssd_chunk2(
    const float* __restrict__ xbc, const float* __restrict__ acum_ws,
    const float* __restrict__ states, const float* __restrict__ Dp,
    float* __restrict__ y)
{
  __shared__ v16bf sCe[8][4][32];   // exp(a_cum)*C as A-frags
  __shared__ v16bf sSf[4][4][32];   // S^T as B-frags (k=n, col=p)
  __shared__ float sAc[128];

  const int tid = threadIdx.x;
  const int bx  = blockIdx.x;
  const int h   = bx & 63;
  const int c   = (bx >> 6) % NCHUNK;
  const int b   = bx / (64 * NCHUNK);
  const int l0  = c * CHUNK;
  const float* xc = xbc + (size_t)(b * SEQ + l0) * CONV_DIM;
  __bf16* pCe = (__bf16*)sCe;
  __bf16* pS  = (__bf16*)sSf;

  if (tid < 128)
    sAc[tid] = acum_ws[(((size_t)(b * NCHUNK + c)) * N_HEADS + h) * 128 + tid];
  __syncthreads();
  {
    const int lB = tid >> 5, n4 = (tid & 31) << 2;
#pragma unroll
    for (int u = 0; u < 16; ++u) {
      int l = lB + 8 * u;
      float e = expf(sAc[l]);
      float4 cv = *(const float4*)(xc + (size_t)l * CONV_DIM + (INNER + D_STATE) + n4);
      cv.x *= e; cv.y *= e; cv.z *= e; cv.w *= e;
      stage_a4(pCe, l, n4, cv, 4);
    }
  }
  size_t sbase = (((size_t)(b * NCHUNK + c)) * N_HEADS + h) * (HEAD_DIM * D_STATE);
  {
    const int pB = tid >> 5, n4 = (tid & 31) << 2;
#pragma unroll
    for (int u = 0; u < 8; ++u) {
      int p = pB + 8 * u;
      float4 sv = *(const float4*)(states + sbase + (size_t)p * D_STATE + n4);
      stage_b4(pS, n4, p, sv, 4);
    }
  }
  __syncthreads();

  const int wid = tid >> 5, lane = tid & 31;
  const int g = lane >> 4, r = lane & 15;

  v8f acc[4] = {};
#pragma unroll
  for (int ko = 0; ko < 4; ++ko) {
    v16bf af = sCe[wid][ko][lane];
#pragma unroll
    for (int j = 0; j < 4; ++j)
      acc[j] = WMMA_BF16(af, sSf[j][ko][lane], acc[j]);
  }
  float dh = Dp[h];
#pragma unroll
  for (int j = 0; j < 4; ++j)
#pragma unroll
    for (int q = 0; q < 8; ++q) {
      int l = wid * 16 + q + 8 * g;
      int p = j * 16 + r;
      size_t oi = (size_t)(b * SEQ + l0 + l) * INNER + h * HEAD_DIM + p;
      y[oi] = y[oi] + acc[j][q] + dh * xc[(size_t)l * CONV_DIM + h * HEAD_DIM + p];
    }
}

// ---------------------------------------------------------------------------
// y *= silu(z); RMS-norm; * norm_w.  One workgroup per token.
// ---------------------------------------------------------------------------
__global__ __launch_bounds__(256) void gate_rmsnorm(
    const float* __restrict__ y, const float* __restrict__ zxbcdt,
    const float* __restrict__ norm_w, float* __restrict__ yn)
{
  __shared__ float red[256];
  const int tid = threadIdx.x;
  const size_t row = blockIdx.x;
  float vals[16];
  float ss = 0.f;
#pragma unroll
  for (int j = 0; j < 16; ++j) {
    int ch = tid + 256 * j;
    float zv = zxbcdt[row * D_IN_PROJ + ch];
    float yv = y[row * INNER + ch] * silu_f(zv);
    vals[j] = yv;
    ss += yv * yv;
  }
  red[tid] = ss;
  __syncthreads();
  for (int s = 128; s > 0; s >>= 1) {
    if (tid < s) red[tid] += red[tid + s];
    __syncthreads();
  }
  float scale = rsqrtf(red[0] / (float)INNER + 1e-6f);
#pragma unroll
  for (int j = 0; j < 16; ++j) {
    int ch = tid + 256 * j;
    yn[row * INNER + ch] = vals[j] * scale * norm_w[ch];
  }
}

// ---------------------------------------------------------------------------
extern "C" void kernel_launch(void* const* d_in, const int* in_sizes, int n_in,
                              void* d_out, int out_size, void* d_ws, size_t ws_size,
                              hipStream_t stream) {
  (void)in_sizes; (void)n_in; (void)out_size; (void)ws_size;
  const float* x       = (const float*)d_in[0];
  const float* W_in    = (const float*)d_in[1];
  const float* b_in    = (const float*)d_in[2];
  const float* conv_w  = (const float*)d_in[3];
  const float* conv_b  = (const float*)d_in[4];
  const float* A_log   = (const float*)d_in[5];
  const float* dt_bias = (const float*)d_in[6];
  const float* D_param = (const float*)d_in[7];
  const float* norm_w  = (const float*)d_in[8];
  const float* W_out   = (const float*)d_in[9];
  const float* b_out   = (const float*)d_in[10];
  float* out = (float*)d_out;

  char* p = (char*)d_ws;
  auto alloc = [&](size_t bytes) -> void* {
    char* q = p;
    p += (bytes + 255) & ~(size_t)255;
    return (void*)q;
  };
  float* zxbcdt = (float*)alloc((size_t)NTOK * D_IN_PROJ * sizeof(float));
  float* xbc    = (float*)alloc((size_t)NTOK * CONV_DIM * sizeof(float));
  float* dtw    = (float*)alloc((size_t)NTOK * N_HEADS * sizeof(float));
  float* adisc  = (float*)alloc((size_t)NTOK * N_HEADS * sizeof(float));
  float* acum   = (float*)alloc((size_t)BATCH * NCHUNK * N_HEADS * CHUNK * sizeof(float));
  float* asum   = (float*)alloc((size_t)BATCH * NCHUNK * N_HEADS * sizeof(float));
  float* states = (float*)alloc((size_t)BATCH * NCHUNK * N_HEADS * HEAD_DIM * D_STATE * sizeof(float));
  float* yb     = (float*)alloc((size_t)NTOK * INNER * sizeof(float));
  float* yn     = (float*)alloc((size_t)NTOK * INNER * sizeof(float));
  __bf16* xPK   = (__bf16*)alloc((size_t)NTOK * N_HIDDEN * sizeof(__bf16));
  __bf16* winPK = (__bf16*)alloc((size_t)WIN_NPAD * N_HIDDEN * sizeof(__bf16));
  __bf16* ynPK  = (__bf16*)alloc((size_t)NTOK * INNER * sizeof(__bf16));
  __bf16* woutPK= (__bf16*)alloc((size_t)N_HIDDEN * INNER * sizeof(__bf16));

  dim3 blk(256);
  // 0) pack GEMM operands to bf16 fragment order (one streaming pass each)
  pack_a_kernel<NTOK, N_HIDDEN>
      <<<dim3(N_HIDDEN / 32, NTOK / 128), blk, 0, stream>>>(x, xPK);
  pack_b_kernel<N_HIDDEN, D_IN_PROJ, WIN_NPAD>
      <<<dim3(WIN_NPAD / 128, N_HIDDEN / 32), blk, 0, stream>>>(W_in, winPK);
  pack_b_kernel<INNER, N_HIDDEN, N_HIDDEN>
      <<<dim3(N_HIDDEN / 128, INNER / 32), blk, 0, stream>>>(W_out, woutPK);
  // 1) in-projection: (4096 x 2048) @ (2048 x 8512)
  gemm_pk<NTOK, D_IN_PROJ, WIN_NPAD, N_HIDDEN>
      <<<dim3(WIN_NPAD / 128, NTOK / 128), blk, 0, stream>>>(
          xPK, winPK, b_in, zxbcdt);
  // 2) dt / a prep
  dt_prep<<<dim3(NTOK * N_HEADS / 256), blk, 0, stream>>>(
      zxbcdt, A_log, dt_bias, dtw, adisc);
  // 3) causal depthwise conv + SiLU
  conv_silu<<<dim3(CONV_DIM / 256, NTOK), blk, 0, stream>>>(
      zxbcdt, conv_w, conv_b, xbc);
  // 4) SSD intra-chunk (y_diag + local states)
  ssd_chunk1<<<dim3(BATCH * NCHUNK * N_HEADS), blk, 0, stream>>>(
      xbc, dtw, adisc, yb, states, acum, asum);
  // 5) inter-chunk state scan
  scan_states<<<dim3(BATCH * N_HEADS), blk, 0, stream>>>(states, asum);
  // 6) SSD cross-chunk contribution + D residual
  ssd_chunk2<<<dim3(BATCH * NCHUNK * N_HEADS), blk, 0, stream>>>(
      xbc, acum, states, D_param, yb);
  // 7) gate + RMS norm
  gate_rmsnorm<<<dim3(NTOK), blk, 0, stream>>>(yb, zxbcdt, norm_w, yn);
  // 8) pack yn, out-projection: (4096 x 4096) @ (4096 x 2048)
  pack_a_kernel<NTOK, INNER>
      <<<dim3(INNER / 32, NTOK / 128), blk, 0, stream>>>(yn, ynPK);
  gemm_pk<NTOK, N_HIDDEN, N_HIDDEN, INNER>
      <<<dim3(N_HIDDEN / 128, NTOK / 128), blk, 0, stream>>>(
          ynPK, woutPK, b_out, out);
}
